// HybridRuleGNN_41566693490983
// MI455X (gfx1250) — matrix operand order repaired and checked by reading, not compiled, not run on hardware
//
// HybridRuleGNN on MI455X (gfx1250, wave32, WMMA + async-to-LDS).
//
// Roofline: classifier GEMM (200000x1026 @ 1026x512 ~ 210 GFLOP) dominates.
// All GEMMs run on matrix cores (v_wmma_f32_16x16x32_f16, f32 accumulate).
// Activations/weights are pre-converted to f16 once (weights also pre-
// transposed to [N][K]) so GEMM tile staging is a pure copy -> issued as
// global_load_async_to_lds_b128 (ASYNCcnt) instead of load+cvt+ds_store.
// Graph phase: xl/xr (51MB each) fit the 192MB L2 so per-edge gathers hit L2;
// scatter via native global f32 atomics; segment-max via order-preserving
// uint atomicMax. edge_repr is never materialized: the fused classifier
// kernel synthesizes [s|d|s*d|s-d|cos|shared] A-fragments from f16 LDS tiles.
#include <hip/hip_runtime.h>
#include <hip/hip_bf16.h>

typedef _Float16 half_t;
typedef __attribute__((ext_vector_type(16))) _Float16 v16h;
typedef __attribute__((ext_vector_type(8)))  _Float16 v8h;
typedef __attribute__((ext_vector_type(8)))  float    v8f;
typedef __attribute__((ext_vector_type(4)))  float    v4f;

#define LRELU 0.2f

// global_load_async_to_lds_b128: VDST = per-lane LDS byte address,
// second operand = per-lane 64-bit global address (GV mode). Tracked by
// ASYNCcnt; completion = data visible in LDS (CDNA5 ISA).
#define ASYNC_CP_B128(ldsaddr, gptr)                                        \
  asm volatile("global_load_async_to_lds_b128 %0, %1, off"                  \
               :: "v"(ldsaddr), "v"(gptr) : "memory")
#define WAIT_ASYNC0() asm volatile("s_wait_asynccnt 0" ::: "memory")

__device__ __forceinline__ unsigned lds_addr_of(const void* p) {
  return (unsigned)(unsigned long long)p;  // low 32 bits of LDS aperture = DS offset
}

// ---- order-preserving float<->uint map for atomicMax on float scores ----
__device__ __forceinline__ unsigned f2ord(float f) {
  unsigned u = __float_as_uint(f);
  return (u & 0x80000000u) ? ~u : (u | 0x80000000u);
}
__device__ __forceinline__ float ord2f(unsigned u) {
  u = (u & 0x80000000u) ? (u & 0x7fffffffu) : ~u;
  return __uint_as_float(u);
}
#define ORD_NEG_INF 0x007fffffu  // f2ord(-inf)

__device__ __forceinline__ float lrelu(float v) { return (v > 0.f) ? v : LRELU * v; }

// =====================================================================
// Precision prep: f32->f16 activation copies; weight transpose+convert.
// =====================================================================
__global__ void cvt_f16_k(const float* __restrict__ in, half_t* __restrict__ out,
                          long long n)
{
  long long i = (long long)blockIdx.x * 256 + threadIdx.x;
  if (i < n) out[i] = (half_t)in[i];
}

// Wt[n][kp] = (k<K) ? W[k*N+n] : 0   (Kp = K rounded up to 32)
__global__ void transpose_cvt_k(const float* __restrict__ W, half_t* __restrict__ Wt,
                                int K, int N, int Kp)
{
  long long i = (long long)blockIdx.x * 256 + threadIdx.x;
  if (i >= (long long)N * Kp) return;
  int n = (int)(i / Kp), k = (int)(i % Kp);
  Wt[i] = (k < K) ? (half_t)W[(size_t)k * N + n] : (half_t)0.0f;
}

// =====================================================================
// GEMM: out[M,N] = act(A[M,K] @ W[K,N] + bias).  A is f16 [M][K] (K%32==0),
// Wt is f16 pre-transposed [N][K].  Block 256 thr = 8 waves; block tile
// 256(M) x 64(N); each wave owns 32x64 = 8 WMMA accumulators per K-chunk.
// Tile staging = pure f16 copy via global_load_async_to_lds_b128.
// =====================================================================
#define GEMM_BM 256
#define GEMM_BN 64
#define GEMM_BK 32

__global__ __launch_bounds__(256) void gemm_f16_wmma(
    const half_t* __restrict__ A, const half_t* __restrict__ Wt,
    const float* __restrict__ bias, float* __restrict__ out,
    int M, int N, int K, int act)
{
  __shared__ alignas(64) half_t Asm[GEMM_BM][GEMM_BK];   // 16 KB
  __shared__ alignas(64) half_t Bsm[GEMM_BN][GEMM_BK];   //  4 KB

  const int tid  = threadIdx.x;
  const int wave = tid >> 5;
  const int lane = tid & 31;
  const int l16  = lane & 15;
  const int lhalf= lane >> 4;
  const int m0   = blockIdx.x * GEMM_BM;
  const int n0   = blockIdx.y * GEMM_BN;

  const unsigned ldsA = lds_addr_of(&Asm[0][0]);
  const unsigned ldsB = lds_addr_of(&Bsm[0][0]);

  v8f acc[2][4] = {};

  for (int k0 = 0; k0 < K; k0 += GEMM_BK) {
    // stage A: 256 rows x 64B, 4 b128 lanes-ops per thread (row clamp, no branch)
    #pragma unroll
    for (int i = 0; i < 4; ++i) {
      int lin = tid + 256 * i;            // 0..1023
      int r = lin >> 2, cq = lin & 3;     // row, 16B chunk in row
      int gm = m0 + r; if (gm >= M) gm = M - 1;   // clamp: rows >= M unused
      ASYNC_CP_B128(ldsA + (unsigned)(r * 64 + cq * 16),
                    A + (size_t)gm * K + k0 + cq * 8);
    }
    // stage B: 64 rows x 64B, 1 b128 per thread
    {
      int r = tid >> 2, cq = tid & 3;
      ASYNC_CP_B128(ldsB + (unsigned)(r * 64 + cq * 16),
                    Wt + (size_t)(n0 + r) * K + k0 + cq * 8);
    }
    WAIT_ASYNC0();
    __syncthreads();

    // A fragments: two 16-row strips for this wave
    v16h af[2];
    #pragma unroll
    for (int u = 0; u < 2; ++u) {
      int r = (wave << 5) + (u << 4) + l16;
      v8h lo = *(const v8h*)&Asm[r][lhalf * 16];
      v8h hi = *(const v8h*)&Asm[r][lhalf * 16 + 8];
      #pragma unroll
      for (int i = 0; i < 8; ++i) { af[u][i] = lo[i]; af[u][8 + i] = hi[i]; }
    }
    #pragma unroll
    for (int t = 0; t < 4; ++t) {
      v16h bf = *(const v16h*)&Bsm[t * 16 + l16][lhalf * 16];
      acc[0][t] = __builtin_amdgcn_wmma_f32_16x16x32_f16(
          false, af[0], false, bf, (short)0, acc[0][t], false, false);
      acc[1][t] = __builtin_amdgcn_wmma_f32_16x16x32_f16(
          false, af[1], false, bf, (short)0, acc[1][t], false, false);
    }
    __syncthreads();
  }

  #pragma unroll
  for (int u = 0; u < 2; ++u) {
    #pragma unroll
    for (int t = 0; t < 4; ++t) {
      int gn = n0 + t * 16 + l16;
      float bv = bias ? bias[gn] : 0.0f;
      #pragma unroll
      for (int i = 0; i < 8; ++i) {
        int gm = m0 + (wave << 5) + (u << 4) + i + lhalf * 8;
        if (gm < M) {
          float v = acc[u][t][i] + bv;
          if (act == 1) v = fmaxf(v, 0.0f);
          out[(size_t)gm * N + gn] = v;
        }
      }
    }
  }
}

// =====================================================================
// GATv2 edge kernels. E2 = E + N (self loops appended: src=dst=e-E).
// One wave per edge; lane owns 8 contiguous channels; head = lane/8.
// =====================================================================
__global__ __launch_bounds__(256) void gat_score_max(
    const float* __restrict__ xl, const float* __restrict__ xr,
    const float* __restrict__ att, const int* __restrict__ ei,
    float* __restrict__ score, unsigned* __restrict__ mbuf, int E, int E2)
{
  int wid = blockIdx.x * 8 + (threadIdx.x >> 5);
  if (wid >= E2) return;
  int lane = threadIdx.x & 31;
  int s, d;
  if (wid < E) { s = ei[wid]; d = ei[E + wid]; } else { s = d = wid - E; }
  int c0 = lane * 8;
  v4f a  = *(const v4f*)(xl + (size_t)s * 256 + c0);
  v4f b  = *(const v4f*)(xl + (size_t)s * 256 + c0 + 4);
  v4f c  = *(const v4f*)(xr + (size_t)d * 256 + c0);
  v4f dd = *(const v4f*)(xr + (size_t)d * 256 + c0 + 4);
  v4f at0 = *(const v4f*)(att + c0);
  v4f at1 = *(const v4f*)(att + c0 + 4);
  float partial = 0.f;
  #pragma unroll
  for (int i = 0; i < 4; ++i) {
    partial += lrelu(a[i] + c[i]) * at0[i] + lrelu(b[i] + dd[i]) * at1[i];
  }
  partial += __shfl_xor(partial, 1, 32);
  partial += __shfl_xor(partial, 2, 32);
  partial += __shfl_xor(partial, 4, 32);
  if ((lane & 7) == 0) {
    int h = lane >> 3;
    score[(size_t)wid * 4 + h] = partial;
    atomicMax(&mbuf[(size_t)d * 4 + h], f2ord(partial));
  }
}

__global__ __launch_bounds__(256) void gat_exp_denom(
    const float* __restrict__ score, const unsigned* __restrict__ mbuf,
    const int* __restrict__ ei, float* __restrict__ p,
    float* __restrict__ denom, int E, int E2)
{
  long long idx = (long long)blockIdx.x * 256 + threadIdx.x; // over E2*4
  if (idx >= (long long)E2 * 4) return;
  int e = (int)(idx >> 2), h = (int)(idx & 3);
  int d = (e < E) ? ei[E + e] : (e - E);
  float mv = ord2f(mbuf[(size_t)d * 4 + h]);
  float pv = __expf(score[idx] - mv);
  p[idx] = pv;
  __hip_atomic_fetch_add(&denom[(size_t)d * 4 + h], pv,
                         __ATOMIC_RELAXED, __HIP_MEMORY_SCOPE_AGENT);
}

__global__ __launch_bounds__(256) void gat_aggregate(
    const float* __restrict__ xl, const float* __restrict__ p,
    const float* __restrict__ denom, const int* __restrict__ ei,
    float* __restrict__ agg, int E, int E2)
{
  int wid = blockIdx.x * 8 + (threadIdx.x >> 5);
  if (wid >= E2) return;
  int lane = threadIdx.x & 31;
  int s, d;
  if (wid < E) { s = ei[wid]; d = ei[E + wid]; } else { s = d = wid - E; }
  int h = lane >> 3;
  float w = p[(size_t)wid * 4 + h] / denom[(size_t)d * 4 + h];
  int c0 = lane * 8;
  v4f a = *(const v4f*)(xl + (size_t)s * 256 + c0);
  v4f b = *(const v4f*)(xl + (size_t)s * 256 + c0 + 4);
  float* dp = agg + (size_t)d * 256 + c0;
  #pragma unroll
  for (int i = 0; i < 4; ++i) {
    __hip_atomic_fetch_add(dp + i,     a[i] * w, __ATOMIC_RELAXED, __HIP_MEMORY_SCOPE_AGENT);
    __hip_atomic_fetch_add(dp + 4 + i, b[i] * w, __ATOMIC_RELAXED, __HIP_MEMORY_SCOPE_AGENT);
  }
}

__global__ void bias_relu_k(const float* __restrict__ a, const float* __restrict__ b,
                            float* __restrict__ o, long long n)
{
  long long i = (long long)blockIdx.x * 256 + threadIdx.x;
  if (i < n) o[i] = fmaxf(a[i] + b[i & 255], 0.0f);
}

__global__ void max3_k(const float* __restrict__ a, const float* __restrict__ b,
                       const float* __restrict__ c, float* __restrict__ o, long long n)
{
  long long i = (long long)blockIdx.x * 256 + threadIdx.x;
  if (i < n) o[i] = fmaxf(a[i], fmaxf(b[i], c[i]));
}

__global__ void fill_f32(float* p, float v, long long n)
{
  long long i = (long long)blockIdx.x * 256 + threadIdx.x;
  if (i < n) p[i] = v;
}
__global__ void fill_u32(unsigned* p, unsigned v, long long n)
{
  long long i = (long long)blockIdx.x * 256 + threadIdx.x;
  if (i < n) p[i] = v;
}

// =====================================================================
// Per-target-edge cosine + shared-device flags (one wave per edge).
// =====================================================================
__global__ __launch_bounds__(256) void edge_cos_shared(
    const float* __restrict__ hf, const int* __restrict__ te,
    const int* __restrict__ devid, float* __restrict__ cosb,
    float* __restrict__ shb, int T)
{
  int wid = blockIdx.x * 8 + (threadIdx.x >> 5);
  if (wid >= T) return;
  int lane = threadIdx.x & 31;
  int si = te[wid], di = te[T + wid];
  int c0 = lane * 8;
  v4f s0 = *(const v4f*)(hf + (size_t)si * 256 + c0);
  v4f s1 = *(const v4f*)(hf + (size_t)si * 256 + c0 + 4);
  v4f d0 = *(const v4f*)(hf + (size_t)di * 256 + c0);
  v4f d1 = *(const v4f*)(hf + (size_t)di * 256 + c0 + 4);
  float dot = 0, ns = 0, nd = 0;
  #pragma unroll
  for (int i = 0; i < 4; ++i) {
    dot += s0[i] * d0[i] + s1[i] * d1[i];
    ns  += s0[i] * s0[i] + s1[i] * s1[i];
    nd  += d0[i] * d0[i] + d1[i] * d1[i];
  }
  #pragma unroll
  for (int o = 1; o < 32; o <<= 1) {
    dot += __shfl_xor(dot, o, 32);
    ns  += __shfl_xor(ns, o, 32);
    nd  += __shfl_xor(nd, o, 32);
  }
  if (lane == 0) {
    float na = fmaxf(sqrtf(ns), 1e-8f), nb = fmaxf(sqrtf(nd), 1e-8f);
    cosb[wid] = dot / (na * nb);
    shb[wid]  = (devid[si] == devid[di]) ? 1.0f : 0.0f;
  }
}

// =====================================================================
// Fused classifier GEMM1: z[T,512] = [s|d|s*d|s-d|cos|shared] @ Wc1 + bc1.
// Block tile 32(M) x 256(N), 8 waves = 2(M) x 4(N).  A-fragments are
// synthesized from f16 LDS s/d tiles; pre-transposed Wc1 K-panels (Kp=1056)
// are async-copied to LDS.  K = 1026 -> 33 chunks of 32.
// =====================================================================
#define CLS_KP 1056

__global__ __launch_bounds__(256) void cls1_wmma(
    const float* __restrict__ hf, const int* __restrict__ te,
    const float* __restrict__ cosb, const float* __restrict__ shb,
    const half_t* __restrict__ WtC1,  // f16 [512][CLS_KP]
    const float* __restrict__ bc1, float* __restrict__ z, int T)
{
  __shared__ alignas(64) half_t Ssm[32][256];   // 16 KB
  __shared__ alignas(64) half_t Dsm[32][256];   // 16 KB
  __shared__ alignas(64) half_t BT[256][32];    // 16 KB
  __shared__ float extra[32][2];                // cos, shared

  const int tid  = threadIdx.x;
  const int wave = tid >> 5;
  const int lane = tid & 31;
  const int l16  = lane & 15;
  const int lhalf= lane >> 4;
  const int t0   = blockIdx.x * 32;
  const int n0   = blockIdx.y * 256;
  const int wm   = wave & 1;      // M strip (16 rows)
  const int wn   = wave >> 1;     // N strip (64 cols)
  const unsigned ldsBT = lds_addr_of(&BT[0][0]);

  // stage s/d tiles (32 rows x 256) as f16 (gather + convert)
  #pragma unroll 4
  for (int i = 0; i < 32; ++i) {
    int lin = tid + 256 * i;                // 0..8191
    int r = lin >> 8, c = lin & 255;
    int gr = t0 + r; if (gr >= T) gr = T - 1;
    int si = te[gr], di = te[T + gr];
    Ssm[r][c] = (half_t)hf[(size_t)si * 256 + c];
    Dsm[r][c] = (half_t)hf[(size_t)di * 256 + c];
  }
  if (tid < 32) {
    int gr = t0 + tid; if (gr >= T) gr = T - 1;
    extra[tid][0] = cosb[gr];
    extra[tid][1] = shb[gr];
  }

  v8f acc[4] = {};
  for (int kc = 0; kc < 33; ++kc) {
    int kb = kc * 32;
    __syncthreads();   // previous BT reads done (also fences s/d staging)
    // async-stage pre-transposed Wc1 panel: 256 rows x 64B
    #pragma unroll
    for (int i = 0; i < 4; ++i) {
      int lin = tid + 256 * i;              // 0..1023
      int nn = lin >> 2, cq = lin & 3;
      ASYNC_CP_B128(ldsBT + (unsigned)(nn * 64 + cq * 16),
                    WtC1 + (size_t)(n0 + nn) * CLS_KP + kb + cq * 8);
    }
    WAIT_ASYNC0();
    __syncthreads();

    // synthesize A fragment for edge_repr columns kb..kb+31
    int r = (wm << 4) + l16;
    v16h afrag;
    #pragma unroll
    for (int e = 0; e < 16; ++e) {
      int koff = lhalf * 16 + e;
      int kk = kb + koff;
      half_t v;
      if      (kk < 256)  v = Ssm[r][kk];
      else if (kk < 512)  v = Dsm[r][kk - 256];
      else if (kk < 768)  { int c = kk - 512; v = (half_t)((float)Ssm[r][c] * (float)Dsm[r][c]); }
      else if (kk < 1024) { int c = kk - 768; v = (half_t)((float)Ssm[r][c] - (float)Dsm[r][c]); }
      else if (kk == 1024) v = (half_t)extra[r][0];
      else if (kk == 1025) v = (half_t)extra[r][1];
      else                 v = (half_t)0.0f;
      afrag[e] = v;
    }
    #pragma unroll
    for (int t = 0; t < 4; ++t) {
      v16h bfrag = *(const v16h*)&BT[(wn << 6) + t * 16 + l16][lhalf * 16];
      acc[t] = __builtin_amdgcn_wmma_f32_16x16x32_f16(
          false, afrag, false, bfrag, (short)0, acc[t], false, false);
    }
  }

  #pragma unroll
  for (int t = 0; t < 4; ++t) {
    int gn = n0 + (wn << 6) + t * 16 + l16;
    float bv = bc1[gn];
    #pragma unroll
    for (int i = 0; i < 8; ++i) {
      int gm = t0 + (wm << 4) + i + lhalf * 8;
      if (gm < T) z[(size_t)gm * 512 + gn] = acc[t][i] + bv;
    }
  }
}

// =====================================================================
// BatchNorm stats over z[T,512] (training stats, biased var), then fused
// BN + LeakyReLU + GEMM2 (512x8) output.
// =====================================================================
__global__ __launch_bounds__(256) void bn_stats(const float* __restrict__ z,
                                                float* __restrict__ sums, int T)
{
  int t = threadIdx.x;
  int r0 = blockIdx.x * 256;
  float s0 = 0, s1 = 0, q0 = 0, q1 = 0;
  int rend = r0 + 256; if (rend > T) rend = T;
  for (int r = r0; r < rend; ++r) {
    float a = z[(size_t)r * 512 + t];
    float b = z[(size_t)r * 512 + 256 + t];
    s0 += a; q0 += a * a; s1 += b; q1 += b * b;
  }
  __hip_atomic_fetch_add(&sums[t],       s0, __ATOMIC_RELAXED, __HIP_MEMORY_SCOPE_AGENT);
  __hip_atomic_fetch_add(&sums[256 + t], s1, __ATOMIC_RELAXED, __HIP_MEMORY_SCOPE_AGENT);
  __hip_atomic_fetch_add(&sums[512 + t], q0, __ATOMIC_RELAXED, __HIP_MEMORY_SCOPE_AGENT);
  __hip_atomic_fetch_add(&sums[768 + t], q1, __ATOMIC_RELAXED, __HIP_MEMORY_SCOPE_AGENT);
}

__global__ void bn_finalize(const float* __restrict__ sums,
                            const float* __restrict__ gamma,
                            const float* __restrict__ beta,
                            float* __restrict__ ss, int T)
{
  int c = blockIdx.x * 256 + threadIdx.x;
  if (c >= 512) return;
  float invT = 1.0f / (float)T;
  float mu  = sums[c] * invT;
  float var = sums[512 + c] * invT - mu * mu;
  float sc  = gamma[c] * rsqrtf(var + 1e-5f);
  ss[c] = sc;
  ss[512 + c] = beta[c] - mu * sc;
}

__global__ __launch_bounds__(256) void cls2_out(
    const float* __restrict__ z, const float* __restrict__ ss,
    const float* __restrict__ Wc2, const float* __restrict__ bc2,
    float* __restrict__ out, int T)
{
  int tid = threadIdx.x;
  int row = blockIdx.x * 32 + (tid >> 3);
  int col = tid & 7;
  if (row >= T) return;
  float acc = bc2[col];
  const float* zr = z + (size_t)row * 512;
  #pragma unroll 4
  for (int k = 0; k < 512; ++k) {
    float zn = lrelu(zr[k] * ss[k] + ss[512 + k]);
    acc += zn * Wc2[k * 8 + col];
  }
  out[(size_t)row * 8 + col] = acc;
}

// =====================================================================
// Host orchestration
// =====================================================================
extern "C" void kernel_launch(void* const* d_in, const int* in_sizes, int n_in,
                              void* d_out, int out_size, void* d_ws, size_t ws_size,
                              hipStream_t stream)
{
  const float* x     = (const float*)d_in[0];
  const int*   ei    = (const int*)d_in[1];
  const int*   te    = (const int*)d_in[2];
  const int*   devid = (const int*)d_in[3];
  const float* Wp    = (const float*)d_in[4];
  const float* bp    = (const float*)d_in[5];
  const float* Wl1   = (const float*)d_in[6];
  const float* bl1   = (const float*)d_in[7];
  const float* Wr1   = (const float*)d_in[8];
  const float* br1   = (const float*)d_in[9];
  const float* att1  = (const float*)d_in[10];
  const float* bias1 = (const float*)d_in[11];
  const float* Wl2   = (const float*)d_in[12];
  const float* bl2   = (const float*)d_in[13];
  const float* Wr2   = (const float*)d_in[14];
  const float* br2   = (const float*)d_in[15];
  const float* att2  = (const float*)d_in[16];
  const float* bias2 = (const float*)d_in[17];
  const float* Wc1   = (const float*)d_in[18];
  const float* bc1   = (const float*)d_in[19];
  const float* gamma = (const float*)d_in[20];
  const float* beta  = (const float*)d_in[21];
  const float* Wc2   = (const float*)d_in[22];
  const float* bc2   = (const float*)d_in[23];

  const int N  = in_sizes[0] / 64;     // 50000
  const int E  = in_sizes[1] / 2;      // 640000
  const int T  = in_sizes[2] / 2;      // 200000
  const int E2 = E + N;                // with self loops

  // workspace layout (float units; half buffers consume n/2 floats)
  float* ws = (float*)d_ws;
  size_t o = 0;
  float* h0    = ws + o; o += (size_t)N * 256;
  float* h1    = ws + o; o += (size_t)N * 256;
  float* h2    = ws + o; o += (size_t)N * 256;
  float* hf    = ws + o; o += (size_t)N * 256;
  float* xl    = ws + o; o += (size_t)N * 256;
  float* xr    = ws + o; o += (size_t)N * 256;
  float* agg   = ws + o; o += (size_t)N * 256;
  float* score = ws + o; o += (size_t)E2 * 4;
  float* pbuf  = ws + o; o += (size_t)E2 * 4;
  unsigned* mbuf = (unsigned*)(ws + o); o += (size_t)N * 4;
  float* denom = ws + o; o += (size_t)N * 4;
  float* cosb  = ws + o; o += (size_t)T;
  float* shb   = ws + o; o += (size_t)T;
  float* sums  = ws + o; o += 1024;
  float* ss    = ws + o; o += 1024;
  half_t* xh   = (half_t*)(ws + o); o += (size_t)N * 32;        // N*64 halfs
  half_t* hh   = (half_t*)(ws + o); o += (size_t)N * 128;       // N*256 halfs
  half_t* WtP  = (half_t*)(ws + o); o += (256 * 64) / 2;
  half_t* Wtl1 = (half_t*)(ws + o); o += (256 * 256) / 2;
  half_t* Wtr1 = (half_t*)(ws + o); o += (256 * 256) / 2;
  half_t* Wtl2 = (half_t*)(ws + o); o += (256 * 256) / 2;
  half_t* Wtr2 = (half_t*)(ws + o); o += (256 * 256) / 2;
  half_t* WtC1 = (half_t*)(ws + o); o += ((size_t)512 * CLS_KP) / 2;
  float* z     = ws + o; o += (size_t)T * 512;

  const long long nh  = (long long)N * 256;
  const long long ne4 = (long long)E2 * 4;

  auto gemm = [&](const half_t* A, const half_t* Wt, const float* b, float* out,
                  int M, int Nn, int K, int act) {
    dim3 g((M + GEMM_BM - 1) / GEMM_BM, Nn / GEMM_BN);
    gemm_f16_wmma<<<g, 256, 0, stream>>>(A, Wt, b, out, M, Nn, K, act);
  };
  auto tcvt = [&](const float* W, half_t* Wt, int K, int Nn, int Kp) {
    long long n = (long long)Nn * Kp;
    transpose_cvt_k<<<(unsigned)((n + 255) / 256), 256, 0, stream>>>(W, Wt, K, Nn, Kp);
  };

  // ---- one-time precision prep ----
  cvt_f16_k<<<(unsigned)((N * 64LL + 255) / 256), 256, 0, stream>>>(x, xh, (long long)N * 64);
  tcvt(Wp,  WtP,  64,  256, 64);
  tcvt(Wl1, Wtl1, 256, 256, 256);
  tcvt(Wr1, Wtr1, 256, 256, 256);
  tcvt(Wl2, Wtl2, 256, 256, 256);
  tcvt(Wr2, Wtr2, 256, 256, 256);
  tcvt(Wc1, WtC1, 1026, 512, CLS_KP);

  auto run_conv = [&](const half_t* hin_h, const half_t* Wtl, const float* bl,
                      const half_t* Wtr, const float* br, const float* att,
                      const float* bias, float* hout) {
    gemm(hin_h, Wtl, bl, xl, N, 256, 256, 0);
    gemm(hin_h, Wtr, br, xr, N, 256, 256, 0);
    fill_u32<<<(unsigned)((N * 4 + 255) / 256), 256, 0, stream>>>(mbuf, ORD_NEG_INF, (long long)N * 4);
    fill_f32<<<(unsigned)((N * 4 + 255) / 256), 256, 0, stream>>>(denom, 0.f, (long long)N * 4);
    fill_f32<<<(unsigned)((nh + 255) / 256), 256, 0, stream>>>(agg, 0.f, nh);
    gat_score_max<<<(E2 + 7) / 8, 256, 0, stream>>>(xl, xr, att, ei, score, mbuf, E, E2);
    gat_exp_denom<<<(unsigned)((ne4 + 255) / 256), 256, 0, stream>>>(score, mbuf, ei, pbuf, denom, E, E2);
    gat_aggregate<<<(E2 + 7) / 8, 256, 0, stream>>>(xl, pbuf, denom, ei, agg, E, E2);
    bias_relu_k<<<(unsigned)((nh + 255) / 256), 256, 0, stream>>>(agg, bias, hout, nh);
  };

  // h0 = relu(x @ Wp + bp)
  gemm(xh, WtP, bp, h0, N, 256, 64, 1);
  // GAT layer 1 (A in f16)
  cvt_f16_k<<<(unsigned)((nh + 255) / 256), 256, 0, stream>>>(h0, hh, nh);
  run_conv(hh, Wtl1, bl1, Wtr1, br1, att1, bias1, h1);
  // GAT layer 2
  cvt_f16_k<<<(unsigned)((nh + 255) / 256), 256, 0, stream>>>(h1, hh, nh);
  run_conv(hh, Wtl2, bl2, Wtr2, br2, att2, bias2, h2);
  // h_final = max(h0, h1, h2)
  max3_k<<<(unsigned)((nh + 255) / 256), 256, 0, stream>>>(h0, h1, h2, hf, nh);

  // classifier
  edge_cos_shared<<<(T + 7) / 8, 256, 0, stream>>>(hf, te, devid, cosb, shb, T);
  dim3 gc((T + 31) / 32, 2);
  cls1_wmma<<<gc, 256, 0, stream>>>(hf, te, cosb, shb, WtC1, bc1, z, T);
  fill_f32<<<4, 256, 0, stream>>>(sums, 0.f, 1024);
  bn_stats<<<(T + 255) / 256, 256, 0, stream>>>(z, sums, T);
  bn_finalize<<<2, 256, 0, stream>>>(sums, gamma, beta, ss, T);
  cls2_out<<<(T + 31) / 32, 256, 0, stream>>>(z, ss, Wc2, bc2, (float*)d_out, T);
}
